// AdditiveAttention_2276332667237
// MI455X (gfx1250) — matrix operand order repaired and checked by reading, not compiled, not run on hardware
//
#include <hip/hip_runtime.h>
#include <hip/hip_fp16.h>

#define B_ 32
#define S_ 8192
#define D_ 256
#define Q_ 256
#define U_ 128

typedef __attribute__((ext_vector_type(16))) _Float16 v16h;
typedef __attribute__((ext_vector_type(8)))  _Float16 v8h;
typedef __attribute__((ext_vector_type(8)))  float    v8f;
typedef __attribute__((ext_vector_type(4)))  float    v4f;
typedef __attribute__((ext_vector_type(4)))  unsigned u32x4;
typedef __attribute__((ext_vector_type(8)))  int      i32x8;
typedef __attribute__((ext_vector_type(4)))  int      i32x4;

#if __has_builtin(__builtin_amdgcn_tanhf)
#define TANHF(x) __builtin_amdgcn_tanhf(x)
#else
#define TANHF(x) tanhf(x)
#endif

// LDS pitch for staged W1 rows: 256 halves + 8 pad halves = 264 (528 B, 16B-aligned)
#define W1_PITCH 264
#define QN 32            // N-rows per staged quarter tile

// ---------------------------------------------------------------------------
// TDM: stage a QN x 256-half tile of W1t into LDS with pad_interval=128 dw
// (512B) / pad_amount=4 dw (16B) -> LDS pitch 264 halves. Wave-uniform args.
// ---------------------------------------------------------------------------
#if __has_builtin(__builtin_amdgcn_tensor_load_to_lds)
__device__ __forceinline__ void tdm_load_w1(const _Float16* gsrc, unsigned lds_addr) {
    unsigned long long gaddr = (unsigned long long)(const void*)gsrc;

    u32x4 g0;
    g0[0] = 1u;                                    // count=1 valid D#
    g0[1] = lds_addr;                              // LDS dest (bytes)
    g0[2] = (unsigned)(gaddr & 0xffffffffull);     // global addr lo
    g0[3] = (unsigned)((gaddr >> 32) & 0x01ffffffu) | (2u << 30); // type=2

    i32x8 g1;
    // data_size=1 (2B), pad_enable, pad_interval=6 (128 dw), pad_amount=3 (4 dw)
    g1[0] = (1 << 16) | (1 << 20) | (6 << 22) | (3 << 25);
    g1[1] = (int)(256u << 16);                     // tensor_dim0 = 256
    g1[2] = (int)((unsigned)QN << 16);             // tensor_dim1 = 32
    g1[3] = (int)(256u << 16);                     // tile_dim0 = 256
    g1[4] = QN;                                    // tile_dim1 = 32
    g1[5] = 256;                                   // tensor_dim0_stride = 256
    g1[6] = (int)(8192u << 16);                    // tensor_dim1_stride lo
    g1[7] = 0;
    i32x4 gz4 = {0, 0, 0, 0};
    i32x8 gz8 = {0, 0, 0, 0, 0, 0, 0, 0};
    __builtin_amdgcn_tensor_load_to_lds(g0, g1, gz4, gz4, gz8, 0);
}
#endif

// ---------------------------------------------------------------------------
// Kernel A: prep (32 blocks). Block b computes proj_q[b,:]; blocks also slice
// the W1 transpose->f16 and zero the context output.
// ---------------------------------------------------------------------------
__global__ __launch_bounds__(256)
void prep_kernel(const float* __restrict__ query,
                 const float* __restrict__ W1,
                 const float* __restrict__ W2,
                 _Float16* __restrict__ W1t,
                 float* __restrict__ pq,
                 float* __restrict__ ctx_out) {
    __shared__ float red[256];
    const int blk = blockIdx.x;          // 0..31 (= batch index)
    const int t   = threadIdx.x;

    // proj_q[blk,u]: 256 threads = 128 u-columns x 2 q-halves
    const int u = t & 127;
    const int h = t >> 7;
    const float* qr = query + blk * Q_ + h * 128;
    float acc = 0.0f;
    for (int q = 0; q < 128; ++q)
        acc = fmaf(qr[q], W2[(h * 128 + q) * U_ + u], acc);
    red[t] = acc;
    __syncthreads();
    if (h == 0) pq[blk * U_ + u] = red[u] + red[u + 128];

    // W1t[n*256 + k] = (f16) W1[k*128 + n], sliced: 1024 elems per block
    for (int i = blk * 1024 + t; i < blk * 1024 + 1024; i += 256) {
        const int n = i >> 8;
        const int k = i & 255;
        W1t[i] = (_Float16)W1[k * U_ + n];
    }

    // zero context accumulator: 8192 = 32 blocks * 256 threads
    ctx_out[blk * 256 + t] = 0.0f;
}

// ---------------------------------------------------------------------------
// Kernel B: scores[b,s] = V^T tanh(values[b,s,:] @ W1 + proj_q[b,:])
// Grid (S/128, B), 8 waves/block; each wave owns a 16-row s strip, keeps all
// 8 A-fragments resident, and W1 is TDM double-buffered in 32-row quarters.
// ---------------------------------------------------------------------------
__global__ __launch_bounds__(256)
void score_kernel(const float* __restrict__ values,
                  const _Float16* __restrict__ W1t,   // [N=128][K=256] f16
                  const float* __restrict__ pq,       // [B][U]
                  const float* __restrict__ V,        // [U]
                  float* __restrict__ scores) {       // [B][S]
    __shared__ _Float16 lds_w1[2][QN * W1_PITCH];     // 2 x 16896 B

    const int b    = blockIdx.y;
    const int s0   = blockIdx.x * 128 + (threadIdx.x >> 5) * 16;
    const int lane = threadIdx.x & 31;
    const int nlo  = lane & 15;
    const int g    = lane >> 4;

    float pqr[8], vr[8];
#pragma unroll
    for (int nt = 0; nt < 8; ++nt) {
        pqr[nt] = pq[b * U_ + nt * 16 + nlo];
        vr[nt]  = V[nt * 16 + nlo];
    }

    // ---- load all 8 A fragments for this strip (values read once) ----
    const float* arow = values + ((size_t)b * S_ + (size_t)(s0 + nlo)) * D_;
    v16h a[8];
#pragma unroll
    for (int kk = 0; kk < 8; ++kk) {
        const float* p = arow + kk * 32 + g * 16;
        v4f f0 = *(const v4f*)(p);
        v4f f1 = *(const v4f*)(p + 4);
        v4f f2 = *(const v4f*)(p + 8);
        v4f f3 = *(const v4f*)(p + 12);
#pragma unroll
        for (int j = 0; j < 4; ++j) {
            a[kk][j]      = (_Float16)f0[j];
            a[kk][4 + j]  = (_Float16)f1[j];
            a[kk][8 + j]  = (_Float16)f2[j];
            a[kk][12 + j] = (_Float16)f3[j];
        }
    }

    float sc[8];
#pragma unroll
    for (int r = 0; r < 8; ++r) sc[r] = 0.0f;

#if __has_builtin(__builtin_amdgcn_tensor_load_to_lds)
    const unsigned lds0 = (unsigned)(unsigned long long)(void*)&lds_w1[0][0];
    const unsigned lds1 = (unsigned)(unsigned long long)(void*)&lds_w1[1][0];
    if (threadIdx.x < 32) tdm_load_w1(W1t, lds0);   // quarter 0 -> buf 0
#endif

#pragma unroll
    for (int q = 0; q < 4; ++q) {
        const int buf = q & 1;
#if __has_builtin(__builtin_amdgcn_tensor_load_to_lds)
        if (threadIdx.x < 32) {
            if (q < 3) {  // prefetch next quarter into the other buffer
                tdm_load_w1(W1t + (size_t)(q + 1) * QN * D_, (q & 1) ? lds0 : lds1);
#if __has_builtin(__builtin_amdgcn_s_wait_tensorcnt)
                __builtin_amdgcn_s_wait_tensorcnt(1);
#endif
            } else {
#if __has_builtin(__builtin_amdgcn_s_wait_tensorcnt)
                __builtin_amdgcn_s_wait_tensorcnt(0);
#endif
            }
        }
#else
        for (int i = threadIdx.x; i < QN * 32; i += 256) {   // b128 chunks
            const int n  = i >> 5;
            const int kc = (i & 31) * 8;
            *(v8h*)&lds_w1[buf][n * W1_PITCH + kc] =
                *(const v8h*)&W1t[(size_t)(q * QN + n) * D_ + kc];
        }
#endif
        __syncthreads();   // publish buf; all waves aligned

        v8f c0 = {}, c1 = {};
#pragma unroll
        for (int kk = 0; kk < 8; ++kk) {
            const _Float16* wp0 = &lds_w1[buf][nlo * W1_PITCH + kk * 32 + g * 16];
            const _Float16* wp1 = wp0 + 16 * W1_PITCH;
            v8h b0 = *(const v8h*)(wp0);
            v8h b1 = *(const v8h*)(wp0 + 8);
            v8h b2 = *(const v8h*)(wp1);
            v8h b3 = *(const v8h*)(wp1 + 8);
            v16h bb0 = __builtin_shufflevector(b0, b1,
                0, 1, 2, 3, 4, 5, 6, 7, 8, 9, 10, 11, 12, 13, 14, 15);
            v16h bb1 = __builtin_shufflevector(b2, b3,
                0, 1, 2, 3, 4, 5, 6, 7, 8, 9, 10, 11, 12, 13, 14, 15);
            c0 = __builtin_amdgcn_wmma_f32_16x16x32_f16(
                false, a[kk], false, bb0, (short)0, c0, false, false);
            c1 = __builtin_amdgcn_wmma_f32_16x16x32_f16(
                false, a[kk], false, bb1, (short)0, c1, false, false);
        }

        // epilogue: columns N = (2q)*16+nlo and (2q+1)*16+nlo
#pragma unroll
        for (int r = 0; r < 8; ++r) {
            float t0 = TANHF(c0[r] + pqr[2 * q]);
            float t1 = TANHF(c1[r] + pqr[2 * q + 1]);
            sc[r] = fmaf(t0, vr[2 * q], sc[r]);
            sc[r] = fmaf(t1, vr[2 * q + 1], sc[r]);
        }
        __syncthreads();   // release buf before it is re-staged
    }

    // reduce over the 16 lanes of each half (sums the N dimension)
#pragma unroll
    for (int r = 0; r < 8; ++r) {
        float v = sc[r];
        v += __shfl_xor(v, 1, 32);
        v += __shfl_xor(v, 2, 32);
        v += __shfl_xor(v, 4, 32);
        v += __shfl_xor(v, 8, 32);
        if (nlo == 0) scores[(size_t)b * S_ + s0 + r + 8 * g] = v;
    }
}

// ---------------------------------------------------------------------------
// Kernel C: softmax over S per batch. Grid = B, 256 threads, 32 elems/thread.
// ---------------------------------------------------------------------------
__global__ __launch_bounds__(256)
void softmax_kernel(const float* __restrict__ scores,
                    float* __restrict__ attn) {
    __shared__ float red[256];
    const int b = blockIdx.x;
    const int t = threadIdx.x;
    const float* sr = scores + (size_t)b * S_;

    float x[32];
#pragma unroll
    for (int i = 0; i < 32; ++i) x[i] = sr[t + i * 256];

    float m = -3.402823466e38f;
#pragma unroll
    for (int i = 0; i < 32; ++i) m = fmaxf(m, x[i]);
    red[t] = m;
    __syncthreads();
    for (int off = 128; off > 0; off >>= 1) {
        if (t < off) red[t] = fmaxf(red[t], red[t + off]);
        __syncthreads();
    }
    const float mx = red[0];
    __syncthreads();

    float s = 0.0f;
#pragma unroll
    for (int i = 0; i < 32; ++i) {
        x[i] = __expf(x[i] - mx);
        s += x[i];
    }
    red[t] = s;
    __syncthreads();
    for (int off = 128; off > 0; off >>= 1) {
        if (t < off) red[t] += red[t + off];
        __syncthreads();
    }
    const float inv = 1.0f / red[0];

    float* ar = attn + (size_t)b * S_;
#pragma unroll
    for (int i = 0; i < 32; ++i) ar[t + i * 256] = x[i] * inv;
}

// ---------------------------------------------------------------------------
// Kernel D: context[b,d] = sum_s attn[b,s] * values[b,s,d]
// ---------------------------------------------------------------------------
__global__ __launch_bounds__(256)
void context_kernel(const float* __restrict__ values,
                    const float* __restrict__ attn,
                    float* __restrict__ ctx) {
    const int b     = blockIdx.y;
    const int sBase = blockIdx.x * 256;
    const int d     = threadIdx.x;

    const float* vp = values + ((size_t)b * S_ + (size_t)sBase) * D_ + d;
    const float* ap = attn + (size_t)b * S_ + sBase;

    float acc = 0.0f;
#pragma unroll 4
    for (int i = 0; i < 256; ++i) {
        acc = fmaf(ap[i], vp[(size_t)i * D_], acc);
    }
    __hip_atomic_fetch_add(&ctx[b * D_ + d], acc,
                           __ATOMIC_RELAXED, __HIP_MEMORY_SCOPE_AGENT);
}

// ---------------------------------------------------------------------------
extern "C" void kernel_launch(void* const* d_in, const int* in_sizes, int n_in,
                              void* d_out, int out_size, void* d_ws, size_t ws_size,
                              hipStream_t stream) {
    (void)in_sizes; (void)n_in; (void)out_size; (void)ws_size;

    const float* query  = (const float*)d_in[0];   // (B,Q)
    const float* values = (const float*)d_in[1];   // (B,S,D)
    const float* W1     = (const float*)d_in[2];   // (D,U)
    const float* W2     = (const float*)d_in[3];   // (Q,U)
    const float* V      = (const float*)d_in[4];   // (U,1)

    float* ctx  = (float*)d_out;          // (B,D)
    float* attn = ctx + B_ * D_;          // (B,S,1)

    _Float16* W1t = (_Float16*)d_ws;                                  // 64 KB
    float* pq     = (float*)((char*)d_ws + (size_t)D_ * U_ * 2);      // 16 KB
    float* scores = pq + B_ * U_;                                     // 1 MB

    prep_kernel<<<B_, 256, 0, stream>>>(query, W1, W2, W1t, pq, ctx);
    score_kernel<<<dim3(S_ / 128, B_), 256, 0, stream>>>(values, W1t, pq, V, scores);
    softmax_kernel<<<B_, 256, 0, stream>>>(scores, attn);
    context_kernel<<<dim3(S_ / 256, B_), 256, 0, stream>>>(values, attn, ctx);
}
